// MultiplicativeAttentionWithMask_48412871360746
// MI455X (gfx1250) — compile-verified, matrix operand order checked by reading, not run on hardware
//
#include <hip/hip_runtime.h>
#include <math.h>

#define N_  128
#define T_  2048
#define H_  512
#define O_  512
#define LDS_STRIDE 520   // 512 + 8 u16 pad -> 4-bank rotate per row

typedef float          f32x4  __attribute__((ext_vector_type(4)));
typedef float          f32x8  __attribute__((ext_vector_type(8)));
typedef unsigned short u16x4  __attribute__((ext_vector_type(4)));
typedef unsigned int   u32x4  __attribute__((ext_vector_type(4)));
typedef __bf16         bf16x16 __attribute__((ext_vector_type(16)));

// exact pointee type expected by the async-LDS builtin (per compiler diagnostic)
typedef int b128v __attribute__((vector_size(16)));
typedef __attribute__((address_space(1))) b128v* as1_b128;
typedef __attribute__((address_space(3))) b128v* as3_b128;

#if __has_builtin(__builtin_amdgcn_tanhf)
#define FAST_TANH(x) __builtin_amdgcn_tanhf(x)
#else
#define FAST_TANH(x) tanhf(x)
#endif

#if __has_builtin(__builtin_amdgcn_global_load_async_to_lds_b128) && \
    __has_builtin(__builtin_amdgcn_s_wait_asynccnt)
#define HAVE_ASYNC_LDS 1
#else
#define HAVE_ASYNC_LDS 0
#endif

union Frag {
    bf16x16 v;
    struct { u32x4 lo, hi; } u;
};

static __device__ __forceinline__ unsigned short f32_to_bf16(float f) {
    unsigned int u = __float_as_uint(f);
    unsigned int lsb = (u >> 16) & 1u;
    u += 0x7fffu + lsb;                 // round-to-nearest-even
    return (unsigned short)(u >> 16);
}

static __device__ __forceinline__ void load_frag(Frag& f, const unsigned short* p) {
    f.u.lo = *(const u32x4*)(p);
    f.u.hi = *(const u32x4*)(p + 16);
}

// stage one 16x512 bf16 B-tile (16 KB) into LDS (stride-padded), async if available
static __device__ __forceinline__ void stage_B(const unsigned short* __restrict__ src,
                                               unsigned short* dst, int tid) {
#pragma unroll
    for (int j = 0; j < 4; ++j) {
        int i   = tid + j * 256;        // 1024 x 16-byte chunks
        int row = i >> 6;               // /64 chunks per row
        int col = (i & 63) * 8;
#if HAVE_ASYNC_LDS
        __builtin_amdgcn_global_load_async_to_lds_b128(
            (as1_b128)(src + row * 512 + col),
            (as3_b128)(dst + row * LDS_STRIDE + col), 0, 0);
#else
        *(u32x4*)(dst + row * LDS_STRIDE + col) = *(const u32x4*)(src + row * 512 + col);
#endif
    }
}

// ---------------- Kernel 0: We (f32) -> bf16 scratch ----------------
__global__ void k_cvt_we(const float* __restrict__ We, unsigned short* __restrict__ WeB) {
    int i4 = (blockIdx.x * 256 + threadIdx.x) * 4;
    f32x4 f = *(const f32x4*)(We + i4);
    u16x4 h;
    h.x = f32_to_bf16(f.x); h.y = f32_to_bf16(f.y);
    h.z = f32_to_bf16(f.z); h.w = f32_to_bf16(f.w);
    *(u16x4*)(WeB + i4) = h;
}

// ---------------- Kernel 1: dec_key = tanh(dh @ Wd^T + bd) ----------------
__global__ void k_deckey(const float* __restrict__ dh, const float* __restrict__ Wd,
                         const float* __restrict__ bd, float* __restrict__ deckey) {
    const int n = blockIdx.x, tid = threadIdx.x;
    __shared__ float hrow[H_];
    for (int i = tid; i < H_; i += 256) hrow[i] = dh[n * H_ + i];
    __syncthreads();
    for (int o = tid; o < O_; o += 256) {
        const f32x4* wr = (const f32x4*)(Wd + o * H_);
        float acc = 0.f;
#pragma unroll 4
        for (int h4 = 0; h4 < H_ / 4; ++h4) {
            f32x4 w = wr[h4];
            const float* hh = hrow + h4 * 4;
            acc += hh[0] * w.x + hh[1] * w.y + hh[2] * w.z + hh[3] * w.w;
        }
        deckey[n * O_ + o] = FAST_TANH(acc + bd[o]);
    }
}

// ---------------- Kernel 2: fused enc_keys GEMM (bf16 WMMA) + tanh + dot ----------------
// grid (T/128, N), block 256 (8 waves). A tile (128 t-rows, bf16) in LDS; all 16
// A-fragments held in VGPRs across the o-loop; B o-tiles double-buffered in LDS via
// async-to-LDS staging shared by all 8 waves.
__global__ void __launch_bounds__(256, 1)
k_scores(const float* __restrict__ enc, const unsigned short* __restrict__ WeB,
         const float* __restrict__ be, const float* __restrict__ deckey,
         float* __restrict__ scores) {
    extern __shared__ unsigned short Abuf[];            // 128 x LDS_STRIDE u16
    __shared__ unsigned short Bbuf[2][16 * LDS_STRIDE]; // double-buffered B tile

    const int tid  = threadIdx.x;
    const int n    = blockIdx.y;
    const int t0   = blockIdx.x * 128;
    const int w    = tid >> 5;
    const int lane = tid & 31;

    // kick off async stage of the first B tile; overlaps with A conversion below
    stage_B(WeB, Bbuf[0], tid);

    // --- cooperative load 128x512 f32 -> bf16 LDS tile ---
    const float* Ablk = enc + (size_t)n * T_ * H_ + (size_t)t0 * H_;
    for (int i = tid; i < (128 * H_) / 4; i += 256) {
        int e   = i * 4;
        int row = e >> 9;
        int col = e & (H_ - 1);
        f32x4 f = *(const f32x4*)(Ablk + row * H_ + col);
        u16x4 h;
        h.x = f32_to_bf16(f.x); h.y = f32_to_bf16(f.y);
        h.z = f32_to_bf16(f.z); h.w = f32_to_bf16(f.w);
        *(u16x4*)(Abuf + row * LDS_STRIDE + col) = h;
    }
#if HAVE_ASYNC_LDS
    __builtin_amdgcn_s_wait_asynccnt(0);
#endif
    __syncthreads();

    const int m      = lane & 15;               // A row within tile / B column (o)
    const int kHalf  = (lane < 16) ? 0 : 8;     // 16-bit operand K-half per ISA layout
    const unsigned short* abase = Abuf + (w * 16 + m) * LDS_STRIDE + kHalf;

    // preload ALL 16 A-fragments: o-invariant, kept in VGPRs for the whole kernel
    Frag afr[16];
#pragma unroll
    for (int k = 0; k < H_ / 32; ++k) load_frag(afr[k], abase + k * 32);

    float partial[8] = {0.f, 0.f, 0.f, 0.f, 0.f, 0.f, 0.f, 0.f};

    for (int ot = 0; ot < O_ / 16; ++ot) {
        const int cur = ot & 1;
        if (ot + 2 < O_ / 16)                   // warm L2 two tiles ahead
            __builtin_prefetch(WeB + (ot + 2) * 16 * H_ + tid * 32, 0, 2);
        if (ot + 1 < O_ / 16)                   // async-stage next tile while computing
            stage_B(WeB + (ot + 1) * 16 * H_, Bbuf[cur ^ 1], tid);

        const int o   = ot * 16 + m;
        const float beo = be[o];
        const float dk  = deckey[n * O_ + o];
        const unsigned short* bbase = Bbuf[cur] + m * LDS_STRIDE + kHalf;

        f32x8 acc = {0.f, 0.f, 0.f, 0.f, 0.f, 0.f, 0.f, 0.f};
        Frag b0, b1;
        load_frag(b0, bbase);
#pragma unroll
        for (int k = 0; k < H_ / 32; k += 2) {
            load_frag(b1, bbase + (k + 1) * 32);
            acc = __builtin_amdgcn_wmma_f32_16x16x32_bf16(
                false, afr[k].v, false, b0.v, (short)0, acc, false, false);
            if (k + 2 < H_ / 32)
                load_frag(b0, bbase + (k + 2) * 32);
            acc = __builtin_amdgcn_wmma_f32_16x16x32_bf16(
                false, afr[k + 1].v, false, b1.v, (short)0, acc, false, false);
        }

#pragma unroll
        for (int r = 0; r < 8; ++r)
            partial[r] += FAST_TANH(acc[r] + beo) * dk;   // enc_key * dec_key over o

        if (ot + 1 < O_ / 16) {
#if HAVE_ASYNC_LDS
            __builtin_amdgcn_s_wait_asynccnt(0);          // next tile staged
#endif
            __syncthreads();                              // all waves done with cur buf
        }
    }

    // --- reduce over the 16 o-columns (stay within each 16-lane half) ---
#pragma unroll
    for (int r = 0; r < 8; ++r) {
        float v = partial[r];
        v += __shfl_xor(v, 1);
        v += __shfl_xor(v, 2);
        v += __shfl_xor(v, 4);
        v += __shfl_xor(v, 8);
        partial[r] = v;
    }
    if ((lane & 15) == 0) {
        const int mbase = (lane < 16) ? 0 : 8;   // C layout: lanes 0-15 -> M 0-7, 16-31 -> M 8-15
        float* srow = scores + (size_t)n * T_ + t0 + w * 16 + mbase;
#pragma unroll
        for (int r = 0; r < 8; ++r) srow[r] = partial[r];
    }
}

// ---------------- Kernel 3: masked softmax over T ----------------
__global__ void k_softmax(const float* __restrict__ scores, const float* __restrict__ mask,
                          float* __restrict__ weights) {
    const int n = blockIdx.x, tid = threadIdx.x;
    __shared__ float red[256];
    float loc[8];
    float mx = -INFINITY;
#pragma unroll
    for (int i = 0; i < 8; ++i) {
        loc[i] = scores[n * T_ + tid + i * 256];
        mx = fmaxf(mx, loc[i]);
    }
    red[tid] = mx; __syncthreads();
    for (int s = 128; s > 0; s >>= 1) {
        if (tid < s) red[tid] = fmaxf(red[tid], red[tid + s]);
        __syncthreads();
    }
    mx = red[0]; __syncthreads();

    float sum = 0.f;
#pragma unroll
    for (int i = 0; i < 8; ++i) {
        float e = __expf(loc[i] - mx) * mask[n * T_ + tid + i * 256];
        loc[i] = e; sum += e;
    }
    red[tid] = sum; __syncthreads();
    for (int s = 128; s > 0; s >>= 1) {
        if (tid < s) red[tid] += red[tid + s];
        __syncthreads();
    }
    const float inv = 1.0f / red[0];
#pragma unroll
    for (int i = 0; i < 8; ++i)
        weights[n * T_ + tid + i * 256] = loc[i] * inv;
}

// ---------------- Kernel 4: context[n,h] = sum_t enc[n,t,h] * w[n,t] ----------------
__global__ void k_context(const float* __restrict__ enc, const float* __restrict__ weights,
                          float* __restrict__ out) {
    const int n = blockIdx.x, tid = threadIdx.x;
    const int h = blockIdx.y * 256 + tid;
    __shared__ float wsm[T_];
#if HAVE_ASYNC_LDS
    {
        const float* wsrc = weights + (size_t)n * T_;
        for (int i = tid; i < T_ / 4; i += 256) {   // 512 x b128 async transfers
            __builtin_amdgcn_global_load_async_to_lds_b128(
                (as1_b128)(wsrc + i * 4),
                (as3_b128)(wsm + i * 4),
                0, 0);
        }
        __builtin_amdgcn_s_wait_asynccnt(0);
    }
#else
    for (int i = tid; i < T_; i += 256) wsm[i] = weights[n * T_ + i];
#endif
    __syncthreads();
    const float* ebase = enc + (size_t)n * T_ * H_ + h;
    float acc = 0.f;
#pragma unroll 8
    for (int t = 0; t < T_; ++t)
        acc += ebase[(size_t)t * H_] * wsm[t];
    out[n * H_ + h] = acc;
}

extern "C" void kernel_launch(void* const* d_in, const int* in_sizes, int n_in,
                              void* d_out, int out_size, void* d_ws, size_t ws_size,
                              hipStream_t stream) {
    (void)in_sizes; (void)n_in; (void)out_size; (void)ws_size;
    const float* dh   = (const float*)d_in[0];   // [N,H]
    const float* enc  = (const float*)d_in[1];   // [N,T,H]
    const float* mask = (const float*)d_in[2];   // [N,T]
    const float* We   = (const float*)d_in[3];   // [O,H]
    const float* be   = (const float*)d_in[4];   // [O]
    const float* Wd   = (const float*)d_in[5];   // [O,H]
    const float* bd   = (const float*)d_in[6];   // [O]
    float* out = (float*)d_out;                  // [N,H]

    char* ws = (char*)d_ws;
    unsigned short* WeB     = (unsigned short*)(ws);              // 512 KB bf16 weights
    float*          deckey  = (float*)(ws + 524288);              // 256 KB
    float*          scores  = (float*)(ws + 786432);              // 1 MB
    float*          weights = (float*)(ws + 1835008);             // 1 MB

    const size_t lds2 = 128 * LDS_STRIDE * sizeof(unsigned short); // 133,120 B dynamic
    (void)hipFuncSetAttribute((const void*)k_scores,
                              hipFuncAttributeMaxDynamicSharedMemorySize, (int)lds2);

    k_cvt_we <<<dim3((O_ * H_) / (256 * 4)), dim3(256), 0, stream>>>(We, WeB);
    k_deckey <<<dim3(N_), dim3(256), 0, stream>>>(dh, Wd, bd, deckey);
    k_scores <<<dim3(T_ / 128, N_), dim3(256), lds2, stream>>>(enc, WeB, be, deckey, scores);
    k_softmax<<<dim3(N_), dim3(256), 0, stream>>>(scores, mask, weights);
    k_context<<<dim3(N_, H_ / 256), dim3(256), 0, stream>>>(enc, weights, out);
}
